// MultiHeadAttention_68212670595108
// MI455X (gfx1250) — matrix operand order repaired and checked by reading, not compiled, not run on hardware
//
#include <hip/hip_runtime.h>

// ---------------- problem constants ----------------
#define BATCH 2
#define SEQ   2048
#define DMODEL 1024
#define NHEAD 16
#define HDIM  64
#define BS    (BATCH * SEQ)          // 4096
#define NEGV  (-10000.0f)
#define SM_SCALE 0.125f              // 1/sqrt(64)

typedef __bf16 bf16_t;
typedef __attribute__((ext_vector_type(16))) __bf16 v16bf;
typedef __attribute__((ext_vector_type(8)))  __bf16 v8bf;
typedef __attribute__((ext_vector_type(8)))  float  v8f;
typedef __attribute__((ext_vector_type(4)))  float  v4f;

static __device__ inline v16bf frag_cat(v8bf a, v8bf b) {
    return __builtin_shufflevector(a, b, 0,1,2,3,4,5,6,7,8,9,10,11,12,13,14,15);
}

// WMMA D = A(16x32 bf16) * B(32x16 bf16) + C(16x16 f32)
static __device__ inline v8f wmma_bf16(v16bf a, v16bf b, v8f c) {
    return __builtin_amdgcn_wmma_f32_16x16x32_bf16(false, a, false, b, (short)0, c, false, false);
}

// A-fragment from a row-major 16x32 bf16 tile (rowStride in elements).
static __device__ inline v16bf load_frag_a(const bf16_t* base, int rowStride, int lane) {
    const bf16_t* p = base + (lane & 15) * rowStride + (lane >> 4) * 8;
    return frag_cat(*(const v8bf*)p, *(const v8bf*)(p + 16));
}

// B-fragment from an N-major tile: 16 rows (N) x 32 cols (K).
static __device__ inline v16bf load_frag_b(const bf16_t* base, int rowStride, int lane) {
    const bf16_t* p = base + (lane & 15) * rowStride + (lane >> 4) * 8;
    return frag_cat(*(const v8bf*)p, *(const v8bf*)(p + 16));
}

// pack 8 f32 (two v4f) into v8bf
static __device__ inline v8bf pack_bf8(v4f a, v4f b) {
    v8bf r;
    r[0] = (bf16_t)a.x; r[1] = (bf16_t)a.y; r[2] = (bf16_t)a.z; r[3] = (bf16_t)a.w;
    r[4] = (bf16_t)b.x; r[5] = (bf16_t)b.y; r[6] = (bf16_t)b.z; r[7] = (bf16_t)b.w;
    return r;
}

// ---------------------------------------------------------------------------
// GEMM: Y[m,n] = sum_k A[m,k] * W[n,k]   (X @ W^T), M=4096, N=K=1024 fixed.
// Compile-time flags; double-buffered LDS; 256 thr (8 waves), 128x128 tile,
// K-step 32 (one WMMA K). Global loads for tile k+1 are in flight while the
// WMMA burst consumes tile k.
// ---------------------------------------------------------------------------
template<int A_BF16, int OUT_F32, int SPLIT>
__global__ __launch_bounds__(256)
void gemm_xwt(const float* __restrict__ Xf, const bf16_t* __restrict__ Xb,
              const float* __restrict__ W,
              bf16_t* __restrict__ Yb, float* __restrict__ Yf)
{
    constexpr int K  = DMODEL;
    constexpr int NK = K / 32;                 // 32 k-steps

    __shared__ __align__(16) bf16_t As[2][128 * 32];
    __shared__ __align__(16) bf16_t Bsh[2][128 * 32];

    const int lane = threadIdx.x & 31;
    const int wid  = threadIdx.x >> 5;      // 0..7
    const int wy   = wid & 3;               // M quadrant (32 rows)
    const int wx   = wid >> 2;              // N half (64 cols)
    const int m0   = blockIdx.y * 128;
    const int n0   = blockIdx.x * 128;

    v8f acc[2][4];
    #pragma unroll
    for (int i = 0; i < 2; ++i)
        #pragma unroll
        for (int t = 0; t < 4; ++t) { v8f z = {}; acc[i][t] = z; }

    const int lrow = threadIdx.x >> 1;          // 0..127
    const int lcol = (threadIdx.x & 1) * 16;    // 0 or 16

    const float*  aF = A_BF16 ? nullptr : (Xf + (size_t)(m0 + lrow) * K + lcol);
    const bf16_t* aB = A_BF16 ? (Xb + (size_t)(m0 + lrow) * K + lcol) : nullptr;
    const float*  wF = W + (size_t)(n0 + lrow) * K + lcol;

    // staging registers
    v8bf ar0, ar1;          // A tile chunk (16 bf16)
    v4f  af0, af1, af2, af3;
    v4f  wf0, wf1, wf2, wf3;

    auto fetch = [&](int k0) {
        if constexpr (A_BF16) {
            ar0 = *(const v8bf*)(aB + k0);
            ar1 = *(const v8bf*)(aB + k0 + 8);
        } else {
            af0 = *(const v4f*)(aF + k0);      af1 = *(const v4f*)(aF + k0 + 4);
            af2 = *(const v4f*)(aF + k0 + 8);  af3 = *(const v4f*)(aF + k0 + 12);
        }
        wf0 = *(const v4f*)(wF + k0);      wf1 = *(const v4f*)(wF + k0 + 4);
        wf2 = *(const v4f*)(wF + k0 + 8);  wf3 = *(const v4f*)(wF + k0 + 12);
    };
    auto stage = [&](int buf) {
        bf16_t* da = &As[buf][lrow * 32 + lcol];
        if constexpr (A_BF16) {
            *(v8bf*)da       = ar0;
            *(v8bf*)(da + 8) = ar1;
        } else {
            *(v8bf*)da       = pack_bf8(af0, af1);
            *(v8bf*)(da + 8) = pack_bf8(af2, af3);
        }
        bf16_t* db = &Bsh[buf][lrow * 32 + lcol];
        *(v8bf*)db       = pack_bf8(wf0, wf1);
        *(v8bf*)(db + 8) = pack_bf8(wf2, wf3);
    };

    fetch(0);
    stage(0);
    __syncthreads();

    for (int kt = 0; kt < NK; ++kt) {
        const int cur = kt & 1;
        if (kt + 1 < NK) fetch((kt + 1) * 32);   // next tile in flight

        v16bf fa[2], fb[4];
        #pragma unroll
        for (int i = 0; i < 2; ++i)
            fa[i] = load_frag_a(&As[cur][(wy * 32 + i * 16) * 32], 32, lane);
        #pragma unroll
        for (int t = 0; t < 4; ++t)
            fb[t] = load_frag_b(&Bsh[cur][(wx * 64 + t * 16) * 32], 32, lane);
        #pragma unroll
        for (int i = 0; i < 2; ++i)
            #pragma unroll
            for (int t = 0; t < 4; ++t)
                acc[i][t] = wmma_bf16(fa[i], fb[t], acc[i][t]);

        if (kt + 1 < NK) stage(cur ^ 1);
        __syncthreads();
    }

    // ---- store (C/D layout: lane&15 = col, vgpr j = row j + 8*(lane>>4)) ----
    const int ln = lane & 15, lh = lane >> 4;
    #pragma unroll
    for (int i = 0; i < 2; ++i) {
        #pragma unroll
        for (int t = 0; t < 4; ++t) {
            #pragma unroll
            for (int j = 0; j < 8; ++j) {
                int m = m0 + wy * 32 + i * 16 + j + lh * 8;
                int n = n0 + wx * 64 + t * 16 + ln;
                float v = acc[i][t][j];
                size_t idx;
                if constexpr (SPLIT) {   // (m,n) -> (B,H,S,HD)
                    int b = m >> 11, l = m & (SEQ - 1);
                    int h = n >> 6,  d = n & (HDIM - 1);
                    idx = (((size_t)(b * NHEAD + h)) * SEQ + l) * HDIM + d;
                } else {
                    idx = (size_t)m * DMODEL + n;
                }
                if constexpr (OUT_F32) Yf[idx] = v;
                else                   Yb[idx] = (bf16_t)v;
            }
        }
    }
}

// ---------------------------------------------------------------------------
// Flash attention: one block (4 waves / 128 thr) per (b*H+h, 64-query rows).
// Double-buffered K / V^T tiles of 32 keys; online softmax; WMMA for both
// Q@K^T and P@V. Causal early exit.
// ---------------------------------------------------------------------------
__global__ __launch_bounds__(128)
void flash_attn(const bf16_t* __restrict__ Qh, const bf16_t* __restrict__ Kh,
                const bf16_t* __restrict__ Vh, const int* __restrict__ attn_mask,
                const int* __restrict__ mask_future_p, bf16_t* __restrict__ ctx)
{
    __shared__ __align__(16) bf16_t Ks[2][32 * 64];     // (s, d)
    __shared__ __align__(16) bf16_t Vt[2][64 * 32];     // (d, s) = V^T
    __shared__ __align__(16) bf16_t Ps[4][16 * 32];     // per-wave P tile (l, s)

    const int lane = threadIdx.x & 31;
    const int wid  = threadIdx.x >> 5;      // 0..3
    const int qt   = blockIdx.x;
    const int bh   = blockIdx.y;
    const int b    = bh >> 4;               // / NHEAD
    const int h    = bh & (NHEAD - 1);
    const int mf   = mask_future_p[0];
    const int q0   = qt * 64;
    const int qrow0 = q0 + wid * 16;

    const bf16_t* Qb = Qh + (size_t)bh * SEQ * HDIM;
    const bf16_t* Kb = Kh + (size_t)bh * SEQ * HDIM;
    const bf16_t* Vb = Vh + (size_t)bh * SEQ * HDIM;

    // Q A-fragments straight from global
    v16bf qa[2];
    #pragma unroll
    for (int c = 0; c < 2; ++c) {
        const bf16_t* p = Qb + (size_t)(qrow0 + (lane & 15)) * HDIM + c * 32 + (lane >> 4) * 8;
        qa[c] = frag_cat(*(const v8bf*)p, *(const v8bf*)(p + 16));
    }

    v8f cacc[4];
    #pragma unroll
    for (int dt = 0; dt < 4; ++dt) { v8f z = {}; cacc[dt] = z; }
    float mrun[8], lrun[8];
    #pragma unroll
    for (int j = 0; j < 8; ++j) { mrun[j] = -3.0e38f; lrun[j] = 0.0f; }

    const int kend = mf ? (q0 + 64) : SEQ;  // causal early exit
    const int nkt  = kend / 32;
    const int ls = threadIdx.x >> 2;        // 0..31 : key row
    const int ld = (threadIdx.x & 3) * 16;  // d chunk

    v8bf kr0, kr1, vr0, vr1;                // staged tile registers
    auto fetch_kv = [&](int kb) {
        const bf16_t* ksrc = Kb + (size_t)(kb + ls) * HDIM + ld;
        kr0 = *(const v8bf*)ksrc; kr1 = *(const v8bf*)(ksrc + 8);
        const bf16_t* vsrc = Vb + (size_t)(kb + ls) * HDIM + ld;
        vr0 = *(const v8bf*)vsrc; vr1 = *(const v8bf*)(vsrc + 8);
    };
    auto stage_kv = [&](int buf) {
        *(v8bf*)&Ks[buf][ls * 64 + ld]     = kr0;
        *(v8bf*)&Ks[buf][ls * 64 + ld + 8] = kr1;
        #pragma unroll
        for (int i = 0; i < 8; ++i) {       // transpose V into (d, s)
            Vt[buf][(ld + i) * 32 + ls]     = vr0[i];
            Vt[buf][(ld + 8 + i) * 32 + ls] = vr1[i];
        }
    };

    fetch_kv(0);
    stage_kv(0);
    __syncthreads();

    for (int it = 0; it < nkt; ++it) {
        const int kb  = it * 32;
        const int cur = it & 1;
        if (it + 1 < nkt) {
            fetch_kv(kb + 32);              // next K/V tile in flight
            __builtin_prefetch(Kb + (size_t)(kb + 64 + ls) * HDIM + ld, 0, 1);
            __builtin_prefetch(Vb + (size_t)(kb + 64 + ls) * HDIM + ld, 0, 1);
        }

        // ---- scores: 2 x (16 l x 16 s) tiles, K=64 over 2 chunks ----
        float sc[2][8];
        #pragma unroll
        for (int st = 0; st < 2; ++st) {
            v8f s = {};
            #pragma unroll
            for (int c = 0; c < 2; ++c) {
                const bf16_t* p = &Ks[cur][(st * 16 + (lane & 15)) * 64 + c * 32 + (lane >> 4) * 8];
                v16bf fb = frag_cat(*(const v8bf*)p, *(const v8bf*)(p + 16));
                s = wmma_bf16(qa[c], fb, s);
            }
            const int sg    = kb + st * 16 + (lane & 15);
            const float pad = (attn_mask[b * SEQ + sg] == 0) ? NEGV : 0.0f;
            #pragma unroll
            for (int j = 0; j < 8; ++j) {
                const int lg = qrow0 + j + (lane >> 4) * 8;
                float v = s[j] * SM_SCALE;
                v = (mf && (sg > lg)) ? NEGV : v;
                sc[st][j] = v + pad;
            }
        }

        // ---- online softmax (row reductions across 16-lane half-groups) ----
        #pragma unroll
        for (int j = 0; j < 8; ++j) {
            float t = fmaxf(sc[0][j], sc[1][j]);
            t = fmaxf(t, __shfl_xor(t, 1, 32));
            t = fmaxf(t, __shfl_xor(t, 2, 32));
            t = fmaxf(t, __shfl_xor(t, 4, 32));
            t = fmaxf(t, __shfl_xor(t, 8, 32));
            const float mn    = fmaxf(mrun[j], t);
            const float alpha = __expf(mrun[j] - mn);
            mrun[j] = mn;
            sc[0][j] = __expf(sc[0][j] - mn);
            sc[1][j] = __expf(sc[1][j] - mn);
            float rs = sc[0][j] + sc[1][j];
            rs += __shfl_xor(rs, 1, 32);
            rs += __shfl_xor(rs, 2, 32);
            rs += __shfl_xor(rs, 4, 32);
            rs += __shfl_xor(rs, 8, 32);
            lrun[j] = lrun[j] * alpha + rs;
            #pragma unroll
            for (int dt = 0; dt < 4; ++dt) cacc[dt][j] *= alpha;
        }

        // ---- P (D-layout) -> per-wave LDS -> A-fragment ----
        bf16_t* P = &Ps[wid][0];
        #pragma unroll
        for (int st = 0; st < 2; ++st)
            #pragma unroll
            for (int j = 0; j < 8; ++j)
                P[(j + (lane >> 4) * 8) * 32 + st * 16 + (lane & 15)] = (bf16_t)sc[st][j];
        const bf16_t* pp = P + (lane & 15) * 32 + (lane >> 4) * 8;
        v16bf fp = frag_cat(*(const v8bf*)pp, *(const v8bf*)(pp + 16));

        // ---- ctx += P @ V  (B-fragments from V^T tile) ----
        #pragma unroll
        for (int dt = 0; dt < 4; ++dt) {
            const bf16_t* pv = &Vt[cur][(dt * 16 + (lane & 15)) * 32 + (lane >> 4) * 8];
            v16bf fv = frag_cat(*(const v8bf*)pv, *(const v8bf*)(pv + 16));
            cacc[dt] = wmma_bf16(fp, fv, cacc[dt]);
        }

        if (it + 1 < nkt) stage_kv(cur ^ 1);
        __syncthreads();
    }

    // ---- normalize + store ctx in (B,S,D) bf16 ----
    #pragma unroll
    for (int j = 0; j < 8; ++j) {
        const float inv = (lrun[j] > 0.0f) ? (1.0f / lrun[j]) : 0.0f;
        const int lg = qrow0 + j + (lane >> 4) * 8;
        const size_t base = ((size_t)b * SEQ + lg) * DMODEL + (size_t)h * HDIM;
        #pragma unroll
        for (int dt = 0; dt < 4; ++dt)
            ctx[base + dt * 16 + (lane & 15)] = (bf16_t)(cacc[dt][j] * inv);
    }
}

// ---------------------------------------------------------------------------
extern "C" void kernel_launch(void* const* d_in, const int* in_sizes, int n_in,
                              void* d_out, int out_size, void* d_ws, size_t ws_size,
                              hipStream_t stream)
{
    const float* q  = (const float*)d_in[0];
    const float* k  = (const float*)d_in[1];
    const float* v  = (const float*)d_in[2];
    const int*   am = (const int*)d_in[3];
    const float* Wq = (const float*)d_in[4];
    const float* Wk = (const float*)d_in[5];
    const float* Wv = (const float*)d_in[6];
    const float* Wo = (const float*)d_in[7];
    const int*   mf = (const int*)d_in[8];
    float* out = (float*)d_out;

    const size_t nEl = (size_t)BATCH * NHEAD * SEQ * HDIM;  // 4,194,304
    bf16_t* Qh  = (bf16_t*)d_ws;
    bf16_t* Kh  = Qh + nEl;
    bf16_t* Vh  = Kh + nEl;
    bf16_t* ctx = Vh + nEl;

    dim3 gg(DMODEL / 128, BS / 128);   // (8, 32)
    // input projections -> (B,H,S,HD) bf16
    gemm_xwt<0,0,1><<<gg, 256, 0, stream>>>(q, nullptr, Wq, Qh, nullptr);
    gemm_xwt<0,0,1><<<gg, 256, 0, stream>>>(k, nullptr, Wk, Kh, nullptr);
    gemm_xwt<0,0,1><<<gg, 256, 0, stream>>>(v, nullptr, Wv, Vh, nullptr);
    // attention
    flash_attn<<<dim3(SEQ / 64, BATCH * NHEAD), 128, 0, stream>>>(Qh, Kh, Vh, am, mf, ctx);
    // output projection -> f32
    gemm_xwt<1,1,0><<<gg, 256, 0, stream>>>(nullptr, ctx, Wo, nullptr, out);
}
